// Attention_3624952398651
// MI455X (gfx1250) — matrix operand order repaired and checked by reading, not compiled, not run on hardware
//
#include <hip/hip_runtime.h>
#include <hip/hip_bf16.h>
#include <math.h>

// Sinkhorn exp-attention for MI455X (gfx1250, wave32, WMMA).
// B=2 H=8 S=1024 D=64 -> BH=16 heads.
typedef _Float16 v16h __attribute__((ext_vector_type(16)));
typedef float    v8f  __attribute__((ext_vector_type(8)));

#define BH   16
#define S    1024
#define D    64
#define LOGS 6.93147180559945309f   // ln(1024)

// ---------------- workspace layout (bytes) ----------------
// A      : fp32 [16][1024][1024]           64 MB   @ 0
// Ph     : f16  [16][1024][1024]           32 MB   @ 64 MB
// Qh     : f16  [16][1024][64]              2 MB   @ 96 MB
// Kh     : f16  [16][1024][64]              2 MB   @ 98 MB
// Vt     : f16  [16][64][1024] (transposed) 2 MB   @ 100 MB
// u      : fp32 [16][1024]                 64 KB   @ 102 MB
// v      : fp32 [16][1024]                 64 KB   @ 102 MB + 64 KB
// colpart: fp32 [16 chunks][16][1024]       1 MB   @ 102 MB + 128 KB
static constexpr size_t OFF_A   = 0;
static constexpr size_t OFF_PH  = (size_t)64 << 20;
static constexpr size_t OFF_QH  = (size_t)96 << 20;
static constexpr size_t OFF_KH  = (size_t)98 << 20;
static constexpr size_t OFF_VT  = (size_t)100 << 20;
static constexpr size_t OFF_U   = (size_t)102 << 20;
static constexpr size_t OFF_V   = OFF_U + (64 << 10);
static constexpr size_t OFF_CP  = OFF_V + (64 << 10);

// ---------------- fp32 -> f16 conversion (+ V transpose) ----------------
__global__ void sk_cvt_kernel(const float* __restrict__ Q,
                              const float* __restrict__ K,
                              const float* __restrict__ V,
                              _Float16* __restrict__ Qh,
                              _Float16* __restrict__ Kh,
                              _Float16* __restrict__ Vt) {
    int gid = blockIdx.x * blockDim.x + threadIdx.x;   // 0 .. 16*1024*64-1
    Qh[gid] = (_Float16)Q[gid];
    Kh[gid] = (_Float16)K[gid];
    int bh = gid >> 16;            // / (1024*64)
    int r  = gid & 65535;
    int s  = r >> 6;
    int d  = r & 63;
    Vt[((size_t)bh << 16) + ((size_t)d << 10) + s] = (_Float16)V[gid];
}

// ---------------- A = -exp(Q K^T) / 8 via v_wmma_f32_16x16x32_f16 ----------------
// One wave computes one 16x16 tile; 16 heads * 64 * 64 tiles = 65536 waves.
__global__ void sk_qkt_kernel(const _Float16* __restrict__ Qh,
                              const _Float16* __restrict__ Kh,
                              float* __restrict__ A) {
    int wid  = blockIdx.x * (blockDim.x >> 5) + (threadIdx.x >> 5);
    int lane = threadIdx.x & 31;
    int lo   = lane & 15, hi = lane >> 4;

    int hd = wid >> 12;
    int t  = wid & 4095;
    int ti = t >> 6, tj = t & 63;

    const _Float16* qrow = Qh + (((size_t)(hd << 10) + (ti << 4) + lo) << 6);
    const _Float16* krow = Kh + (((size_t)(hd << 10) + (tj << 4) + lo) << 6);

    v8f c = {};
    #pragma unroll
    for (int kc = 0; kc < 2; ++kc) {
        int kb = kc << 5;
        v16h a, b;
        #pragma unroll
        for (int e = 0; e < 8; ++e) {
            a[e]     = qrow[kb + (hi << 3) + e];          // K = 32c + 8h + e
            a[e + 8] = qrow[kb + 16 + (hi << 3) + e];     // K = 32c + 16 + 8h + e
        }
        #pragma unroll
        for (int e = 0; e < 8; ++e) {
            b[e]     = krow[kb + (hi << 3) + e];          // K = 32c + 8h + e
            b[e + 8] = krow[kb + 16 + (hi << 3) + e];     // K = 32c + 16 + 8h + e
        }
        c = __builtin_amdgcn_wmma_f32_16x16x32_f16(false, a, false, b,
                                                   (short)0, c, false, false);
    }

    float* Ah = A + ((size_t)hd << 20);
    int col = (tj << 4) + lo;
    #pragma unroll
    for (int r = 0; r < 8; ++r) {
        int row = (ti << 4) + r + (hi << 3);
        Ah[((size_t)row << 10) + col] = -__expf(c[r]) * 0.125f;
    }
}

// ---------------- u,v = 0 ----------------
__global__ void sk_init_kernel(float* __restrict__ u, float* __restrict__ v) {
    int gid = blockIdx.x * blockDim.x + threadIdx.x;   // 0 .. 16383
    u[gid] = 0.0f;
    v[gid] = 0.0f;
}

// ---------------- u[i] = -lnS - ln( sum_j exp(A[i,j] + v[j]) ) ----------------
// One wave per row (args bounded in [-14, 0.3] -> max-free LSE is safe).
__global__ void sk_row_lse_kernel(const float* __restrict__ A,
                                  const float* __restrict__ v,
                                  float* __restrict__ u) {
    int wid  = blockIdx.x * (blockDim.x >> 5) + (threadIdx.x >> 5);  // global row, 0..16383
    int lane = threadIdx.x & 31;
    int hd   = wid >> 10;
    const float* Arow = A + ((size_t)wid << 10);
    const float* vh   = v + (hd << 10);
    float s = 0.0f;
    #pragma unroll 4
    for (int j = lane; j < S; j += 32)
        s += __expf(Arow[j] + vh[j]);
    #pragma unroll
    for (int off = 16; off; off >>= 1)
        s += __shfl_xor(s, off, 32);
    if (lane == 0)
        u[wid] = -LOGS - __logf(s);
}

// ---------------- column partial sums over 64-row chunks (coalesced, no atomics) ----------------
__global__ void sk_col_part_kernel(const float* __restrict__ A,
                                   const float* __restrict__ u,
                                   float* __restrict__ colpart) {
    int j     = threadIdx.x;        // 0..1023: column
    int chunk = blockIdx.x;         // 0..15
    int hd    = blockIdx.y;         // 0..15
    const float* Ah = A + ((size_t)hd << 20);
    const float* uh = u + (hd << 10);
    int i0 = chunk << 6;
    float s = 0.0f;
    #pragma unroll 4
    for (int r = 0; r < 64; ++r) {
        int i = i0 + r;
        s += __expf(Ah[((size_t)i << 10) + j] + uh[i]);
    }
    colpart[(((size_t)chunk << 4) + hd) * S + j] = s;
}

// ---------------- v[j] = -lnS - ln( sum_chunks ) ----------------
__global__ void sk_col_fin_kernel(const float* __restrict__ colpart,
                                  float* __restrict__ v) {
    int gid = blockIdx.x * blockDim.x + threadIdx.x;   // 0..16383 = hd*1024 + j
    int hd  = gid >> 10;
    int j   = gid & 1023;
    float s = 0.0f;
    #pragma unroll
    for (int c = 0; c < 16; ++c)
        s += colpart[(((size_t)c << 4) + hd) * S + j];
    v[gid] = -LOGS - __logf(s);
}

// ---------------- p_attn = 1024 * exp(A + u + v) ; fp32 -> d_out, f16 -> Ph ----------------
__global__ void sk_pattn_kernel(const float* __restrict__ A,
                                const float* __restrict__ u,
                                const float* __restrict__ v,
                                float* __restrict__ pout,
                                _Float16* __restrict__ Ph) {
    size_t gid = (size_t)blockIdx.x * blockDim.x + threadIdx.x;  // 0 .. 16M-1
    int bh = (int)(gid >> 20);
    int i  = (int)((gid >> 10) & 1023);
    int j  = (int)(gid & 1023);
    float p = __expf(A[gid] + u[(bh << 10) + i] + v[(bh << 10) + j]) * 1024.0f;
    pout[gid] = p;
    Ph[gid]   = (_Float16)p;
}

// ---------------- out = p_attn @ V via v_wmma_f32_16x16x32_f16 ----------------
// One wave per 16x16 out tile; 16 heads * 64 * 4 tiles = 4096 waves, K=1024 (32 steps).
__global__ void sk_pv_kernel(const _Float16* __restrict__ Ph,
                             const _Float16* __restrict__ Vt,
                             float* __restrict__ out) {
    int wid  = blockIdx.x * (blockDim.x >> 5) + (threadIdx.x >> 5);
    int lane = threadIdx.x & 31;
    int lo   = lane & 15, hi = lane >> 4;

    int hd = wid >> 8;
    int t  = wid & 255;
    int ti = t >> 2, tj = t & 3;

    const _Float16* prow = Ph + ((size_t)hd << 20) + (((size_t)(ti << 4) + lo) << 10);
    const _Float16* vrow = Vt + (((size_t)(hd << 6) + (tj << 4) + lo) << 10);

    v8f c = {};
    for (int kc = 0; kc < 32; ++kc) {
        int kb = kc << 5;
        v16h a, b;
        #pragma unroll
        for (int e = 0; e < 8; ++e) {
            a[e]     = prow[kb + (hi << 3) + e];
            a[e + 8] = prow[kb + 16 + (hi << 3) + e];
        }
        #pragma unroll
        for (int e = 0; e < 8; ++e) {
            b[e]     = vrow[kb + (hi << 3) + e];
            b[e + 8] = vrow[kb + 16 + (hi << 3) + e];
        }
        c = __builtin_amdgcn_wmma_f32_16x16x32_f16(false, a, false, b,
                                                   (short)0, c, false, false);
    }

    int col = (tj << 4) + lo;
    #pragma unroll
    for (int r = 0; r < 8; ++r) {
        int row = (ti << 4) + r + (hi << 3);
        out[(((size_t)(hd << 10) + row) << 6) + col] = c[r];
    }
}

extern "C" void kernel_launch(void* const* d_in, const int* in_sizes, int n_in,
                              void* d_out, int out_size, void* d_ws, size_t ws_size,
                              hipStream_t stream) {
    const float* Q = (const float*)d_in[0];
    const float* K = (const float*)d_in[1];
    const float* V = (const float*)d_in[2];

    char* ws = (char*)d_ws;
    float*    A   = (float*)(ws + OFF_A);
    _Float16* Ph  = (_Float16*)(ws + OFF_PH);
    _Float16* Qh  = (_Float16*)(ws + OFF_QH);
    _Float16* Kh  = (_Float16*)(ws + OFF_KH);
    _Float16* Vt  = (_Float16*)(ws + OFF_VT);
    float*    u   = (float*)(ws + OFF_U);
    float*    v   = (float*)(ws + OFF_V);
    float*    cp  = (float*)(ws + OFF_CP);

    float* out  = (float*)d_out;                       // [16][1024][64]
    float* pout = (float*)d_out + (size_t)BH * S * D;  // [16][1024][1024]

    // 1) fp32 -> f16 (+ V transpose): 1,048,576 elements
    sk_cvt_kernel<<<4096, 256, 0, stream>>>(Q, K, V, Qh, Kh, Vt);

    // 2) A = -exp(QK^T)/8 : 65536 waves / 8 waves per block
    sk_qkt_kernel<<<8192, 256, 0, stream>>>(Qh, Kh, A);

    // 3) u = v = 0
    sk_init_kernel<<<64, 256, 0, stream>>>(u, v);

    // 4) 10 Sinkhorn iterations (eps=1 collapsed form)
    for (int it = 0; it < 10; ++it) {
        sk_row_lse_kernel<<<2048, 256, 0, stream>>>(A, v, u);
        sk_col_part_kernel<<<dim3(16, 16), 1024, 0, stream>>>(A, u, cp);
        sk_col_fin_kernel<<<64, 256, 0, stream>>>(cp, v);
    }

    // 5) p_attn (fp32 to output, f16 for the PV GEMM)
    sk_pattn_kernel<<<65536, 256, 0, stream>>>(A, u, v, pout, Ph);

    // 6) out = p_attn @ V : 4096 waves / 8 per block
    sk_pv_kernel<<<512, 256, 0, stream>>>(Ph, Vt, out);
}
